// EfficientAttention_34256659153602
// MI455X (gfx1250) — matrix operand order repaired and checked by reading, not compiled, not run on hardware
//
#include <hip/hip_runtime.h>

// ---------------------------------------------------------------------------
// EfficientAttention for MI455X (gfx1250): bf16 WMMA everywhere.
// B=8, N=4096, C=KC=VC=512, H=8, hk=hv=64.
// Round 2: LDS-staged weight tiles + A-fragment double buffering in the four
// 32768x512x512 GEMMs (cuts L2 weight traffic 8x, overlaps loads with WMMA).
// ---------------------------------------------------------------------------

typedef __attribute__((ext_vector_type(16))) __bf16 v16bf;
typedef __attribute__((ext_vector_type(8)))  float  v8f;

union FragU  { v16bf v; uint4 q[2]; };
union Pack8U { uint4 q; unsigned short h[8]; };

#define DEVINL static __device__ __forceinline__

// LDS weight tile: 64 rows (out-channels) x 512 K, padded to 520 shorts/row
// (1040 B pitch -> 4-bank stagger per row, conflict-free b128 fragment reads).
#define WPITCH 520

DEVINL unsigned short f2bf(float f) {               // round-to-nearest-even
  unsigned int u = __builtin_bit_cast(unsigned int, f);
  u += 0x7FFFu + ((u >> 16) & 1u);
  return (unsigned short)(u >> 16);
}
DEVINL float bf2f(unsigned short h) {
  unsigned int u = ((unsigned int)h) << 16;
  return __builtin_bit_cast(float, u);
}

DEVINL v8f wmma_bf16(v16bf a, v16bf b, v8f c) {
  // D = A(16x32) * B(32x16) + C, f32 accumulate
  return __builtin_amdgcn_wmma_f32_16x16x32_bf16(false, a, false, b, (short)0, c,
                                                 false, false);
}

// A fragment: 16x32 bf16, A row-major (lda in elements).
// lanes 0-15: row m, K=k0..k0+15 ; lanes 16-31: row m, K=k0+16..k0+31
DEVINL v16bf load_a_frag(const unsigned short* __restrict__ A, int lda,
                         int m0, int k0, int lane) {
  const int hl = lane >> 4;
  const int m  = m0 + (lane & 15);
  const unsigned short* p = A + (size_t)m * lda + k0 + hl * 16;
  FragU f;
  f.q[0] = *(const uint4*)(p);
  f.q[1] = *(const uint4*)(p + 8);
  return f.v;
}

// B fragment: 32x16 bf16 from K-innermost storage Bt[n][k] (ldb in elements).
// lanes 0-15: col n, K=k0..k0+15 ; lanes 16-31: col n, K=k0+16..k0+31
DEVINL v16bf load_bt_frag(const unsigned short* __restrict__ Bt, int ldb,
                          int n0, int k0, int lane) {
  const int hl = lane >> 4;
  const int n  = n0 + (lane & 15);
  const unsigned short* p = Bt + (size_t)n * ldb + k0 + hl * 16;
  FragU f;
  f.q[0] = *(const uint4*)(p);
  f.q[1] = *(const uint4*)(p + 8);
  return f.v;
}

// ---------------------------------------------------------------------------
// Prep: xp = bf16(input + pos), inb = bf16(input)
// ---------------------------------------------------------------------------
__global__ __launch_bounds__(256) void prep_inputs(const float* __restrict__ in,
                                                   const float* __restrict__ pos,
                                                   unsigned short* __restrict__ xpb,
                                                   unsigned short* __restrict__ inb,
                                                   int n) {
  for (int i = blockIdx.x * 256 + threadIdx.x; i < n; i += gridDim.x * 256) {
    float a = in[i];
    xpb[i] = f2bf(a + pos[i]);
    inb[i] = f2bf(a);
  }
}

// Wt[n][k] = bf16(W[k][n]), 512x512
__global__ __launch_bounds__(256) void prep_weightT(const float* __restrict__ W,
                                                    unsigned short* __restrict__ Wt) {
  const int id = blockIdx.x * 256 + threadIdx.x;   // 0..262143
  const int n = id & 511, k = id >> 9;
  Wt[(size_t)n * 512 + k] = f2bf(W[(size_t)k * 512 + n]);
}

// ---------------------------------------------------------------------------
// 32768 x 512 GEMM (K=512): out = A(bf16) @ Wt^T + bias.
// MODE 0: bf16 token-major out[m*512+n]            (queries)
// MODE 1: bf16 channel-major out[(b*512+n)*4096+t] (keys/values)
// MODE 2: f32 token-major                          (final output)
// grid (256, 8), block 256 (8 waves). Block tile 128(M) x 64(N).
// Weight tile staged in LDS once per block; A fragments double-buffered.
// ---------------------------------------------------------------------------
template <int MODE>
__global__ __launch_bounds__(256) void gemm512(const unsigned short* __restrict__ A,
                                               const unsigned short* __restrict__ Wt,
                                               const float* __restrict__ bias,
                                               void* __restrict__ outv) {
  __shared__ unsigned short lWt[64 * WPITCH];      // 66,560 B

  const int lane = threadIdx.x & 31;
  const int wave = threadIdx.x >> 5;
  const int m0 = (blockIdx.x * 8 + wave) * 16;
  const int n0 = blockIdx.y * 64;

  // Cooperative stage of the 64x512 weight tile (rows n0..n0+63 contiguous).
  {
    const unsigned short* Wtg = Wt + (size_t)n0 * 512;
    for (int c = threadIdx.x; c < 4096; c += 256) {       // 4096 x 16B chunks
      const int rw = c >> 6, cl = c & 63;
      *(uint4*)(&lWt[rw * WPITCH + cl * 8]) =
          *(const uint4*)(Wtg + (size_t)rw * 512 + cl * 8);
    }
  }
  __syncthreads();

  v8f acc[4];
#pragma unroll
  for (int j = 0; j < 4; ++j)
#pragma unroll
    for (int r = 0; r < 8; ++r) acc[j][r] = 0.f;

  v16bf a = load_a_frag(A, 512, m0, 0, lane);
  for (int k0 = 0; k0 < 512; k0 += 32) {
    // Prefetch next A fragment from global while WMMAs consume LDS B frags.
    v16bf an = load_a_frag(A, 512, m0, (k0 + 32) & 511, lane);
#pragma unroll
    for (int j = 0; j < 4; ++j)
      acc[j] = wmma_bf16(a, load_bt_frag(lWt, WPITCH, 16 * j, k0, lane), acc[j]);
    a = an;
  }

  const int hl = lane >> 4, col = lane & 15;
  if (MODE == 1) {
    unsigned short* out = (unsigned short*)outv;
    const int b   = m0 >> 12;
    const int tok = (m0 & 4095) + hl * 8;
#pragma unroll
    for (int j = 0; j < 4; ++j) {
      const int n = n0 + 16 * j + col;
      const float bv = bias[n];
      Pack8U pk;
#pragma unroll
      for (int r = 0; r < 8; ++r) pk.h[r] = f2bf(acc[j][r] + bv);
      *(uint4*)(out + ((size_t)(b * 512 + n)) * 4096 + tok) = pk.q;
    }
  } else if (MODE == 0) {
    unsigned short* out = (unsigned short*)outv;
#pragma unroll
    for (int j = 0; j < 4; ++j) {
      const int n = n0 + 16 * j + col;
      const float bv = bias[n];
#pragma unroll
      for (int r = 0; r < 8; ++r)
        out[(size_t)(m0 + hl * 8 + r) * 512 + n] = f2bf(acc[j][r] + bv);
    }
  } else {
    float* out = (float*)outv;
#pragma unroll
    for (int j = 0; j < 4; ++j) {
      const int n = n0 + 16 * j + col;
      const float bv = bias[n];
#pragma unroll
      for (int r = 0; r < 8; ++r)
        out[(size_t)(m0 + hl * 8 + r) * 512 + n] = acc[j][r] + bv;
    }
  }
}

// ---------------------------------------------------------------------------
// Key softmax over N=4096 tokens, keys channel-major (B*512 rows), in place.
// ---------------------------------------------------------------------------
__global__ __launch_bounds__(256) void key_softmax(unsigned short* __restrict__ keys) {
  __shared__ float red[256];
  const int t = threadIdx.x;
  unsigned short* row = keys + (size_t)blockIdx.x * 4096;

  float x[16];
  float m = -3.4e38f;
#pragma unroll
  for (int i = 0; i < 16; ++i) { x[i] = bf2f(row[t + i * 256]); m = fmaxf(m, x[i]); }
  red[t] = m; __syncthreads();
  for (int s = 128; s; s >>= 1) { if (t < s) red[t] = fmaxf(red[t], red[t + s]); __syncthreads(); }
  m = red[0]; __syncthreads();

  float sum = 0.f;
#pragma unroll
  for (int i = 0; i < 16; ++i) { x[i] = __expf(x[i] - m); sum += x[i]; }
  red[t] = sum; __syncthreads();
  for (int s = 128; s; s >>= 1) { if (t < s) red[t] += red[t + s]; __syncthreads(); }
  const float inv = 1.f / red[0];

#pragma unroll
  for (int i = 0; i < 16; ++i) row[t + i * 256] = f2bf(x[i] * inv);
}

// ---------------------------------------------------------------------------
// Query softmax over 64 head-channels, queries token-major, in place.
// One thread per (b, n, h). 262144 threads total.
// ---------------------------------------------------------------------------
__global__ __launch_bounds__(256) void query_softmax(unsigned short* __restrict__ q) {
  const int tid = blockIdx.x * 256 + threadIdx.x;                  // 0..262143
  unsigned short* p = q + (size_t)(tid >> 3) * 512 + (size_t)(tid & 7) * 64;

  Pack8U v[8];
  float m = -3.4e38f;
#pragma unroll
  for (int i = 0; i < 8; ++i) {
    v[i].q = *(const uint4*)(p + i * 8);
#pragma unroll
    for (int e = 0; e < 8; ++e) m = fmaxf(m, bf2f(v[i].h[e]));
  }
  float sum = 0.f;
#pragma unroll
  for (int i = 0; i < 8; ++i)
#pragma unroll
    for (int e = 0; e < 8; ++e) sum += __expf(bf2f(v[i].h[e]) - m);
  const float inv = 1.f / sum;
#pragma unroll
  for (int i = 0; i < 8; ++i) {
    Pack8U o;
#pragma unroll
    for (int e = 0; e < 8; ++e) o.h[e] = f2bf(__expf(bf2f(v[i].h[e]) - m) * inv);
    *(uint4*)(p + i * 8) = o.q;
  }
}

// ---------------------------------------------------------------------------
// context[k][v] = sum_n ksm[k][n] * vcm[v][n]  per (b,h); K-dim = 4096 tokens.
// ksm, vcm channel-major (B,512,4096). Output ctxT (B*H, 64 v, 64 k) bf16.
// grid 64 (=B*H), block 512 (16 waves), wave -> one 16x16 tile.
// ---------------------------------------------------------------------------
__global__ __launch_bounds__(512) void context_gemm(const unsigned short* __restrict__ ksm,
                                                    const unsigned short* __restrict__ vcm,
                                                    unsigned short* __restrict__ ctxT) {
  const int lane = threadIdx.x & 31;
  const int wave = threadIdx.x >> 5;            // 0..15
  const int bh = blockIdx.x;                    // 0..63
  const unsigned short* Ah = ksm + (size_t)bh * 64 * 4096;   // rows = k-chan
  const unsigned short* Bh = vcm + (size_t)bh * 64 * 4096;   // Bt[v][n]
  const int kk0 = (wave >> 2) * 16;
  const int v0  = (wave & 3) * 16;

  v8f acc;
#pragma unroll
  for (int r = 0; r < 8; ++r) acc[r] = 0.f;

  for (int n = 0; n < 4096; n += 32)
    acc = wmma_bf16(load_a_frag(Ah, 4096, kk0, n, lane),
                    load_bt_frag(Bh, 4096, v0, n, lane), acc);

  const int hl = lane >> 4, col = lane & 15;
  const int v = v0 + col;
  Pack8U pk;
#pragma unroll
  for (int r = 0; r < 8; ++r) pk.h[r] = f2bf(acc[r]);
  *(uint4*)(ctxT + ((size_t)bh * 64 + v) * 64 + kk0 + hl * 8) = pk.q;
}

// ---------------------------------------------------------------------------
// attended[v][n] = sum_k ctxT[v][k] * qsm[n][k]  per (b,h); write agg token-major.
// grid 2048 (bh*32 + strip-group), block 256 (8 waves).
// ---------------------------------------------------------------------------
__global__ __launch_bounds__(256) void attended_gemm(const unsigned short* __restrict__ ctxT,
                                                     const unsigned short* __restrict__ qsm,
                                                     unsigned short* __restrict__ agg) {
  const int lane = threadIdx.x & 31;
  const int wave = threadIdx.x >> 5;            // 0..7
  const int bh  = blockIdx.x >> 5;              // 0..63
  const int b = bh >> 3, h = bh & 7;
  const int grp = blockIdx.x & 31;
  const int v0 = (wave & 3) * 16;
  const int n0 = (grp * 2 + (wave >> 2)) * 64;

  const unsigned short* Ah = ctxT + (size_t)bh * 4096;              // (64 v x 64 k)
  const unsigned short* Bq = qsm + (size_t)b * 4096 * 512 + h * 64; // Bt[n][k], ldb=512

  v8f acc[4];
#pragma unroll
  for (int j = 0; j < 4; ++j)
#pragma unroll
    for (int r = 0; r < 8; ++r) acc[j][r] = 0.f;

#pragma unroll
  for (int kk = 0; kk < 64; kk += 32) {
    v16bf a = load_a_frag(Ah, 64, v0, kk, lane);
#pragma unroll
    for (int j = 0; j < 4; ++j)
      acc[j] = wmma_bf16(a, load_bt_frag(Bq, 512, n0 + 16 * j, kk, lane), acc[j]);
  }

  const int hl = lane >> 4, col = lane & 15;
#pragma unroll
  for (int j = 0; j < 4; ++j) {
    const int n = n0 + 16 * j + col;
    Pack8U pk;
#pragma unroll
    for (int r = 0; r < 8; ++r) pk.h[r] = f2bf(acc[j][r]);
    *(uint4*)(agg + ((size_t)(b * 4096 + n)) * 512 + h * 64 + v0 + hl * 8) = pk.q;
  }
}

// ---------------------------------------------------------------------------
extern "C" void kernel_launch(void* const* d_in, const int* in_sizes, int n_in,
                              void* d_out, int out_size, void* d_ws, size_t ws_size,
                              hipStream_t stream) {
  (void)in_sizes; (void)n_in; (void)out_size; (void)ws_size;

  const float* input_ = (const float*)d_in[0];
  const float* pos    = (const float*)d_in[1];
  const float* Wk     = (const float*)d_in[2];
  const float* bk     = (const float*)d_in[3];
  const float* Wq     = (const float*)d_in[4];
  const float* bq     = (const float*)d_in[5];
  const float* Wv     = (const float*)d_in[6];
  const float* bv     = (const float*)d_in[7];
  const float* Wr     = (const float*)d_in[8];
  const float* br     = (const float*)d_in[9];
  float* out = (float*)d_out;

  char* ws = (char*)d_ws;
  size_t off = 0;
  auto take = [&](size_t bytes) -> char* {
    char* p = ws + off;
    off += (bytes + 255) & ~(size_t)255;
    return p;
  };
  const size_t BNC2 = (size_t)8 * 4096 * 512 * 2;     // 33.5 MB bf16 tensor
  unsigned short* xpb   = (unsigned short*)take(BNC2);       // xp bf16 (B,N,C)
  unsigned short* inb   = (unsigned short*)take(BNC2);       // input bf16 (B,N,C)
  unsigned short* wkt   = (unsigned short*)take(512 * 512 * 2);
  unsigned short* wqt   = (unsigned short*)take(512 * 512 * 2);
  unsigned short* wvt   = (unsigned short*)take(512 * 512 * 2);
  unsigned short* wrt   = (unsigned short*)take(512 * 512 * 2);
  unsigned short* keysb = (unsigned short*)take(BNC2);       // (B,512,N) ch-major
  unsigned short* qb    = (unsigned short*)take(BNC2);       // (B,N,512) tok-major
  unsigned short* vb    = (unsigned short*)take(BNC2);       // (B,512,N) ch-major
  unsigned short* ctxtb = (unsigned short*)take((size_t)64 * 64 * 64 * 2);
  unsigned short* aggb  = (unsigned short*)take(BNC2);       // (B,N,512) tok-major

  const int nElem = 8 * 4096 * 512;

  prep_inputs<<<4096, 256, 0, stream>>>(input_, pos, xpb, inb, nElem);
  prep_weightT<<<1024, 256, 0, stream>>>(Wk, wkt);
  prep_weightT<<<1024, 256, 0, stream>>>(Wq, wqt);
  prep_weightT<<<1024, 256, 0, stream>>>(Wv, wvt);
  prep_weightT<<<1024, 256, 0, stream>>>(Wr, wrt);

  dim3 gemm_grid(256, 8);
  gemm512<1><<<gemm_grid, 256, 0, stream>>>(xpb, wkt, bk, keysb);  // keys ch-major
  gemm512<0><<<gemm_grid, 256, 0, stream>>>(xpb, wqt, bq, qb);     // queries tok-major
  gemm512<1><<<gemm_grid, 256, 0, stream>>>(inb, wvt, bv, vb);     // values ch-major

  key_softmax<<<4096, 256, 0, stream>>>(keysb);
  query_softmax<<<1024, 256, 0, stream>>>(qb);

  context_gemm<<<64, 512, 0, stream>>>(keysb, vb, ctxtb);
  attended_gemm<<<2048, 256, 0, stream>>>(ctxtb, qb, aggb);

  gemm512<2><<<gemm_grid, 256, 0, stream>>>(aggb, wrt, br, out);
}
